// lightnet_81621558493644
// MI455X (gfx1250) — compile-verified, matrix operand order checked by reading, not run on hardware
//
#include <hip/hip_runtime.h>
#include <hip/hip_bf16.h>

#define EPSBN 1e-5f
#define NEG_INF (-3.4e38f)

typedef _Float16 h16;
typedef __attribute__((ext_vector_type(16))) _Float16 v16h;
typedef __attribute__((ext_vector_type(8)))  _Float16 v8h;
typedef __attribute__((ext_vector_type(8)))  float    v8f;
typedef __attribute__((ext_vector_type(4)))  unsigned int v4u;
typedef __attribute__((ext_vector_type(8)))  int v8i;
typedef __attribute__((ext_vector_type(4)))  int v4i;

static __device__ __forceinline__ int laneId() { return (int)(threadIdx.x & 31u); }

// Load one 16(M/N) x 32(K) f16 WMMA operand tile from a row-major [rows x ld]
// matrix (contiguous K). Per CDNA5 ISA 16-bit A layout: lanes 0-15 hold
// K {0..7,16..23}, lanes 16-31 hold K {8..15,24..31} for rows 0..15.
// The same loader serves B when B^T (N x K) is row-major. Works for global or LDS.
static __device__ __forceinline__ v16h load_tile(const h16* __restrict__ base,
                                                 int row0, int k0, int ld) {
  const int lane = laneId();
  const int r  = row0 + (lane & 15);
  const int kb = k0 + ((lane & 16) ? 8 : 0);
  const h16* p = base + (size_t)r * (size_t)ld + kb;
  union { v16h v; v8h h[2]; } u;
  u.h[0] = *(const v8h*)(p);
  u.h[1] = *(const v8h*)(p + 16);
  return u.v;
}

static __device__ __forceinline__ v8f wmma_acc(v16h a, v16h b, v8f c) {
  return __builtin_amdgcn_wmma_f32_16x16x32_f16(false, a, false, b, (short)0, c,
                                                false, false);
}

// C/D 16x16 f32 tile: lane&15 = column, rows = r + (lane>=16 ? 8 : 0).
static __device__ __forceinline__ void store_tile_f32(float* __restrict__ base,
                                                      int row0, int col0, int ld, v8f c) {
  const int lane = laneId();
  const int col = col0 + (lane & 15);
  const int rb  = row0 + ((lane & 16) ? 8 : 0);
#pragma unroll
  for (int r = 0; r < 8; ++r) base[(size_t)(rb + r) * (size_t)ld + col] = c[r];
}

static __device__ __forceinline__ void store_tile_f16(h16* __restrict__ base,
                                                      int row0, int col0, int ld, v8f c) {
  const int lane = laneId();
  const int col = col0 + (lane & 15);
  const int rb  = row0 + ((lane & 16) ? 8 : 0);
#pragma unroll
  for (int r = 0; r < 8; ++r) base[(size_t)(rb + r) * (size_t)ld + col] = (h16)c[r];
}

// ---------------------------------------------------------------------------
// Stage a 64-row x 128-col f16 panel (row stride 1024 elements in global) into
// LDS via the Tensor Data Mover. Issued by one wave; completion tracked with
// TENSORcnt. Fallback: cooperative copy by the issuing wave.
static __device__ __forceinline__ void stage_panel_tdm(const h16* __restrict__ gsrc,
                                                       h16* __restrict__ ldst) {
#if defined(__gfx1250__) && __has_builtin(__builtin_amdgcn_tensor_load_to_lds)
  unsigned long long ga = (unsigned long long)(uintptr_t)gsrc;
  unsigned int lds_addr = (unsigned int)(uintptr_t)ldst;  // flat LDS addr: low 32 bits
  v4u g0;
  g0[0] = 1u;                                   // count=1 user descriptor
  g0[1] = lds_addr;                             // D#.lds_addr
  g0[2] = (unsigned int)(ga & 0xFFFFFFFFu);     // global_addr[31:0]
  g0[3] = (unsigned int)((ga >> 32) & 0x01FFFFFFu) | 0x80000000u;  // addr[56:32], type=2
  v8i g1;
  g1[0] = (int)(1u << 16);                      // data_size=1 (2 bytes)
  g1[1] = (int)(1024u << 16);                   // tensor_dim0 = 1024 (low 16)
  g1[2] = (int)(64u << 16);                     // tensor_dim0 hi=0, tensor_dim1=64 (low16)
  g1[3] = (int)(128u << 16);                    // tensor_dim1 hi=0, tile_dim0 = 128
  g1[4] = 64;                                   // tile_dim1 = 64, tile_dim2 = 0
  g1[5] = 1024;                                 // tensor_dim0_stride = 1024
  g1[6] = 0;
  g1[7] = 0;
  v4i g2 = {0, 0, 0, 0};
  v4i g3 = {0, 0, 0, 0};
#if __clang_major__ >= 23
  v8i g4 = {0, 0, 0, 0, 0, 0, 0, 0};
  __builtin_amdgcn_tensor_load_to_lds(g0, g1, g2, g3, g4, 0);
#else
  __builtin_amdgcn_tensor_load_to_lds(g0, g1, g2, g3, 0);
#endif
#else
  const int lane = laneId();
  for (int t = lane; t < 64 * 16; t += 32) {
    int row = t >> 4, c8 = (t & 15) * 8;
    *(v8h*)(ldst + row * 128 + c8) = *(const v8h*)(gsrc + (size_t)row * 1024 + c8);
  }
#endif
}

// s_wait_tensorcnt requires a constant-integer immediate: two fixed variants.
static __device__ __forceinline__ void wait_tensorcnt_le1() {
#if defined(__gfx1250__) && __has_builtin(__builtin_amdgcn_s_wait_tensorcnt)
  __builtin_amdgcn_s_wait_tensorcnt(1);
#endif
}
static __device__ __forceinline__ void wait_tensorcnt_0() {
#if defined(__gfx1250__) && __has_builtin(__builtin_amdgcn_s_wait_tensorcnt)
  __builtin_amdgcn_s_wait_tensorcnt(0);
#endif
}

// ---------------------------------------------------------------------------
// f32 -> f16 conversion (weights)
__global__ void k_cvt(const float* __restrict__ s, h16* __restrict__ d, int n) {
  int i = blockIdx.x * 256 + threadIdx.x;
  if (i < n) d[i] = (h16)s[i];
}

// ---------------------------------------------------------------------------
// Position encoding: pos[b,i,n] = max_j x[b,j,n] * (1 - softmax_n(|x[b,j,:]-x[b,i,:]|))
// grid = 48 (b*3), block = 256
__global__ void k_pos(const float* __restrict__ x, float* __restrict__ pos) {
  const int b = blockIdx.x / 3;
  const int i = blockIdx.x % 3;
  const int tid = threadIdx.x;
  __shared__ float red[256];
  __shared__ float mj[3], sj[3];
  const float* xb = x + (size_t)b * 3 * 1024;
  const float* xi = xb + (size_t)i * 1024;
  for (int j = 0; j < 3; ++j) {
    const float* xj = xb + (size_t)j * 1024;
    float mx = NEG_INF;
    for (int n = tid; n < 1024; n += 256) mx = fmaxf(mx, fabsf(xj[n] - xi[n]));
    red[tid] = mx; __syncthreads();
    for (int s = 128; s > 0; s >>= 1) { if (tid < s) red[tid] = fmaxf(red[tid], red[tid + s]); __syncthreads(); }
    if (tid == 0) mj[j] = red[0];
    __syncthreads();
    float sm = 0.f;
    for (int n = tid; n < 1024; n += 256) sm += expf(fabsf(xj[n] - xi[n]) - mj[j]);
    red[tid] = sm; __syncthreads();
    for (int s = 128; s > 0; s >>= 1) { if (tid < s) red[tid] += red[tid + s]; __syncthreads(); }
    if (tid == 0) sj[j] = red[0];
    __syncthreads();
  }
  for (int n = tid; n < 1024; n += 256) {
    float best = NEG_INF;
    for (int j = 0; j < 3; ++j) {
      float d = fabsf(xb[j * 1024 + n] - xi[n]);
      float p = expf(d - mj[j]) / sj[j];
      best = fmaxf(best, xb[j * 1024 + n] * (1.f - p));
    }
    pos[((size_t)b * 3 + i) * 1024 + n] = best;
  }
}

// ---------------------------------------------------------------------------
// Stem: concat(x,pos) -> W1+bn1 -> W2+bn2 -> Wt1+bnt1+relu
// Outputs hres f32 (b,N,256) and h0 f16 (b,N,256). grid (16 b, 16 nblk), 256 thr.
__global__ void k_mlp0(const float* __restrict__ x, const float* __restrict__ pos,
                       const float* __restrict__ W1, const float* __restrict__ g1,
                       const float* __restrict__ b1, const float* __restrict__ m1,
                       const float* __restrict__ v1,
                       const float* __restrict__ W2, const float* __restrict__ g2,
                       const float* __restrict__ b2, const float* __restrict__ m2,
                       const float* __restrict__ v2,
                       const float* __restrict__ Wt1, const float* __restrict__ gt,
                       const float* __restrict__ bt, const float* __restrict__ mt,
                       const float* __restrict__ vt,
                       float* __restrict__ hres, h16* __restrict__ h0) {
  const int b = blockIdx.x;
  const int nbase = blockIdx.y * 64;
  const int tid = threadIdx.x;
  __shared__ float sh6[6][64];
  __shared__ float st1[64][64];
  __shared__ h16  st2[256][64];
  for (int t = tid; t < 6 * 64; t += 256) {
    int i = t >> 6, nl = t & 63, n = nbase + nl;
    float val = (i < 3) ? x[((size_t)b * 3 + i) * 1024 + n]
                        : pos[((size_t)b * 3 + (i - 3)) * 1024 + n];
    sh6[i][nl] = val;
  }
  __syncthreads();
  for (int t = tid; t < 64 * 64; t += 256) {
    int o = t >> 6, nl = t & 63;
    float acc = 0.f;
#pragma unroll
    for (int c = 0; c < 6; ++c) acc += W1[o * 6 + c] * sh6[c][nl];
    float sc = g1[o] * rsqrtf(v1[o] + EPSBN);
    st1[o][nl] = acc * sc + (b1[o] - m1[o] * sc);
  }
  __syncthreads();
  for (int t = tid; t < 256 * 64; t += 256) {
    int o = t >> 6, nl = t & 63;
    float acc = 0.f;
    for (int c = 0; c < 64; ++c) acc += W2[o * 64 + c] * st1[c][nl];
    float sc = g2[o] * rsqrtf(v2[o] + EPSBN);
    st2[o][nl] = (h16)(acc * sc + (b2[o] - m2[o] * sc));
  }
  __syncthreads();
  for (int t = tid; t < 256 * 64; t += 256) {
    int o = t >> 6, nl = t & 63;
    float acc = 0.f;
    for (int c = 0; c < 256; ++c) acc += Wt1[o * 256 + c] * (float)st2[c][nl];
    float sc = gt[o] * rsqrtf(vt[o] + EPSBN);
    float val = fmaxf(acc * sc + (bt[o] - mt[o] * sc), 0.f);
    size_t idx = ((size_t)b * 1024 + nbase + nl) * 256 + o;
    hres[idx] = val;
    h0[idx]   = (h16)val;
  }
}

// ---------------------------------------------------------------------------
// xkT[b,n,d] = sum_c in[b,n,c] * Wk[d,c]   (N x 64, K=256). grid (16 b, 16 nblk)
__global__ void __launch_bounds__(256)
k_gemm_xk(const h16* __restrict__ inT, int ldin, const h16* __restrict__ Wk16,
          h16* __restrict__ xkT) {
  const int b = blockIdx.x, nbase = blockIdx.y * 64;
  const int wave = threadIdx.x >> 5;
  const h16* A = inT + (size_t)b * 1024 * (size_t)ldin;
  h16* out = xkT + (size_t)b * 1024 * 64;
#pragma unroll
  for (int i = 0; i < 2; ++i) {
    int ti = wave * 2 + i, nt = ti >> 2, dt = ti & 3;
    v8f c = {};
#pragma unroll
    for (int k0 = 0; k0 < 256; k0 += 32) {
      v16h a = load_tile(A, nbase + nt * 16, k0, ldin);
      v16h bm = load_tile(Wk16, dt * 16, k0, 256);
      c = wmma_acc(a, bm, c);
    }
    store_tile_f16(out, nbase + nt * 16, dt * 16, 64, c);
  }
}

// ---------------------------------------------------------------------------
// xv_cn[b,c,n] = sum_k Wv[c,k] in[b,n,k] + bv[c]. grid (16 b, 4 cblk, 16 nblk)
__global__ void __launch_bounds__(256)
k_gemm_xv(const h16* __restrict__ inT, int ldin, const h16* __restrict__ Wv16,
          const float* __restrict__ bv, h16* __restrict__ xv) {
  const int b = blockIdx.x, cbase = blockIdx.y * 64, nbase = blockIdx.z * 64;
  const int wave = threadIdx.x >> 5;
  const h16* Bt = inT + (size_t)b * 1024 * (size_t)ldin;
  h16* out = xv + (size_t)b * 256 * 1024;
  const int lane = laneId();
#pragma unroll
  for (int i = 0; i < 2; ++i) {
    int ti = wave * 2 + i, ct = ti >> 2, nt = ti & 3;
    v8f c = {};
#pragma unroll
    for (int k0 = 0; k0 < 256; k0 += 32) {
      v16h a = load_tile(Wv16, cbase + ct * 16, k0, 256);
      v16h bm = load_tile(Bt, nbase + nt * 16, k0, ldin);
      c = wmma_acc(a, bm, c);
    }
    int col = nbase + nt * 16 + (lane & 15);
    int rb  = cbase + ct * 16 + ((lane & 16) ? 8 : 0);
#pragma unroll
    for (int r = 0; r < 8; ++r)
      out[(size_t)(rb + r) * 1024 + col] = (h16)(c[r] + bv[rb + r]);
  }
}

// ---------------------------------------------------------------------------
// E[n,m] = sum_d xkT[n,d] xkT[m,d]  (one batch). grid (16 nblk, 16 mblk)
__global__ void __launch_bounds__(256)
k_energy(const h16* __restrict__ xkTb, float* __restrict__ E) {
  const int nb = blockIdx.x * 64, mb = blockIdx.y * 64;
  const int wave = threadIdx.x >> 5;
#pragma unroll
  for (int i = 0; i < 2; ++i) {
    int ti = wave * 2 + i, nt = ti >> 2, mt = ti & 3;
    v8f c = {};
#pragma unroll
    for (int k0 = 0; k0 < 64; k0 += 32) {
      v16h a = load_tile(xkTb, nb + nt * 16, k0, 64);
      v16h bm = load_tile(xkTb, mb + mt * 16, k0, 64);
      c = wmma_acc(a, bm, c);
    }
    store_tile_f32(E, nb + nt * 16, mb + mt * 16, 1024, c);
  }
}

// Row max / inverse row-sum of exp for softmax. grid 128, 8 waves = 8 rows/block.
__global__ void k_rowstat(const float* __restrict__ E, float* __restrict__ rowM,
                          float* __restrict__ rowZi) {
  const int row = blockIdx.x * 8 + (threadIdx.x >> 5);
  const int lane = laneId();
  const float* r = E + (size_t)row * 1024;
  float mx = NEG_INF;
  for (int j = lane; j < 1024; j += 32) mx = fmaxf(mx, r[j]);
#pragma unroll
  for (int m = 16; m > 0; m >>= 1) mx = fmaxf(mx, __shfl_xor(mx, m, 32));
  float s = 0.f;
  for (int j = lane; j < 1024; j += 32) s += expf(r[j] - mx);
#pragma unroll
  for (int m = 16; m > 0; m >>= 1) s += __shfl_xor(s, m, 32);
  if (lane == 0) { rowM[row] = mx; rowZi[row] = 1.f / s; }
}

// PT[m,n] = exp(E[m,n]-rowM[n]) * rowZi[n]   (valid because E is symmetric).
__global__ void k_ptgen(const float* __restrict__ E, const float* __restrict__ rowM,
                        const float* __restrict__ rowZi, h16* __restrict__ PT) {
  int idx = blockIdx.x * 256 + threadIdx.x;     // grid 4096
  int n = idx & 1023;
  PT[idx] = (h16)(expf(E[idx] - rowM[n]) * rowZi[n]);
}

// ---------------------------------------------------------------------------
// xrT[m,c] = sum_n PT[m,n] xv[c,n]  (one batch). grid (16 mblk, 4 cblk)
// B panel (xv rows cbase..cbase+63) is staged through LDS by the Tensor Data
// Mover, double-buffered in 128-wide K chunks so the DMA overlaps the WMMAs.
__global__ void __launch_bounds__(256)
k_apply(const h16* __restrict__ PT, const h16* __restrict__ xvb,
        h16* __restrict__ xrb) {
  const int mbase = blockIdx.x * 64, cbase = blockIdx.y * 64;
  const int wave = threadIdx.x >> 5;
  const int lane = laneId();
  __shared__ __align__(16) h16 sB[2][64 * 128];

  // wave w handles output tiles (mt, ctA) and (mt, ctA+1): one shared A load.
  const int mt  = wave >> 1;
  const int ctA = (wave & 1) * 2;
  v8f acc0 = {}, acc1 = {};

  const h16* gpanel = xvb + (size_t)cbase * 1024;  // 64 rows, stride 1024

  if (wave == 0) stage_panel_tdm(gpanel, &sB[0][0]);
  for (int ch = 0; ch < 8; ++ch) {
    if (wave == 0) {
      if (ch + 1 < 8) {
        stage_panel_tdm(gpanel + (ch + 1) * 128, &sB[(ch + 1) & 1][0]);
        wait_tensorcnt_le1();   // chunk ch complete; chunk ch+1 still in flight
      } else {
        wait_tensorcnt_0();     // last chunk complete
      }
    }
    __syncthreads();
    const h16* Bl = &sB[ch & 1][0];
#pragma unroll
    for (int kk = 0; kk < 128; kk += 32) {
      const int k0 = ch * 128 + kk;
      if (k0 + 32 < 1024)
        __builtin_prefetch(PT + (size_t)(mbase + mt * 16 + (lane & 15)) * 1024 + k0 + 32, 0, 1);
      v16h a  = load_tile(PT, mbase + mt * 16, k0, 1024);
      v16h b0 = load_tile(Bl, ctA * 16, kk, 128);
      v16h b1 = load_tile(Bl, (ctA + 1) * 16, kk, 128);
      acc0 = wmma_acc(a, b0, acc0);
      acc1 = wmma_acc(a, b1, acc1);
    }
    __syncthreads();
  }
  store_tile_f16(xrb, mbase + mt * 16, cbase + ctA * 16, 256, acc0);
  store_tile_f16(xrb, mbase + mt * 16, cbase + (ctA + 1) * 16, 256, acc1);
}

// ---------------------------------------------------------------------------
// hres += relu(bn(Wt xr + bt)); write f16 into concat slice. grid (16 b, 16 nblk)
__global__ void __launch_bounds__(256)
k_wt_res(const h16* __restrict__ xrT, const h16* __restrict__ Wt16,
         const float* __restrict__ bt, const float* __restrict__ g,
         const float* __restrict__ be, const float* __restrict__ mn,
         const float* __restrict__ vr, float* __restrict__ hres,
         h16* __restrict__ catOut) {
  const int b = blockIdx.x, nbase = blockIdx.y * 64;
  const int wave = threadIdx.x >> 5;
  const int lane = laneId();
  const h16* A = xrT + (size_t)b * 1024 * 256;
#pragma unroll
  for (int i = 0; i < 8; ++i) {
    int ti = wave * 8 + i, nt = ti >> 4, ot = ti & 15;
    v8f c = {};
#pragma unroll
    for (int k0 = 0; k0 < 256; k0 += 32) {
      v16h a = load_tile(A, nbase + nt * 16, k0, 256);
      v16h bm = load_tile(Wt16, ot * 16, k0, 256);
      c = wmma_acc(a, bm, c);
    }
    int o  = ot * 16 + (lane & 15);
    int rb = nbase + nt * 16 + ((lane & 16) ? 8 : 0);
    float sc = g[o] * rsqrtf(vr[o] + EPSBN);
    float sh = be[o] - mn[o] * sc;
    float bias = bt[o];
#pragma unroll
    for (int r = 0; r < 8; ++r) {
      float val = fmaxf((c[r] + bias) * sc + sh, 0.f);
      size_t ridx = (size_t)b * 1024 + rb + r;
      float nv = hres[ridx * 256 + o] + val;
      hres[ridx * 256 + o] = nv;
      catOut[ridx * 1024 + o] = (h16)nv;
    }
  }
}

// ---------------------------------------------------------------------------
// g[b,o] = max_n leaky(bnf(Wf cat)).  grid (16 b, 16 oblk)
__global__ void __launch_bounds__(256)
k_wf(const h16* __restrict__ cat, const h16* __restrict__ Wf16,
     const float* __restrict__ gm, const float* __restrict__ be,
     const float* __restrict__ mn, const float* __restrict__ vr,
     float* __restrict__ gout) {
  const int b = blockIdx.x, obase = blockIdx.y * 64;
  const int wave = threadIdx.x >> 5;
  const int otile = wave >> 1, half = wave & 1;
  const int lane = laneId();
  const h16* Bt = cat + (size_t)b * 1024 * 1024;
  float sc[8], sh[8], vmax[8];
#pragma unroll
  for (int r = 0; r < 8; ++r) {
    int o = obase + otile * 16 + r + ((lane & 16) ? 8 : 0);
    float s = gm[o] * rsqrtf(vr[o] + EPSBN);
    sc[r] = s; sh[r] = be[o] - mn[o] * s; vmax[r] = NEG_INF;
  }
  for (int nt = half * 32; nt < half * 32 + 32; ++nt) {
    v8f c = {};
    for (int k0 = 0; k0 < 1024; k0 += 32) {
      v16h a = load_tile(Wf16, obase + otile * 16, k0, 1024);
      v16h bm = load_tile(Bt, nt * 16, k0, 1024);
      c = wmma_acc(a, bm, c);
    }
#pragma unroll
    for (int r = 0; r < 8; ++r) {
      float val = c[r] * sc[r] + sh[r];
      val = (val > 0.f) ? val : 0.2f * val;
      vmax[r] = fmaxf(vmax[r], val);
    }
  }
#pragma unroll
  for (int m = 1; m < 16; m <<= 1)
#pragma unroll
    for (int r = 0; r < 8; ++r) vmax[r] = fmaxf(vmax[r], __shfl_xor(vmax[r], m, 32));
  __shared__ float part[2][64];
  if ((lane & 15) == 0) {
#pragma unroll
    for (int r = 0; r < 8; ++r)
      part[half][otile * 16 + r + ((lane & 16) ? 8 : 0)] = vmax[r];
  }
  __syncthreads();
  if (threadIdx.x < 64)
    gout[(size_t)b * 1024 + obase + threadIdx.x] =
        fmaxf(part[0][threadIdx.x], part[1][threadIdx.x]);
}

// ---------------------------------------------------------------------------
// Classifier head on (16,1024). grid 16 (one block per batch row).
__global__ void k_head(const float* __restrict__ g,
                       const float* __restrict__ Wl1, const float* __restrict__ g6,
                       const float* __restrict__ b6, const float* __restrict__ m6,
                       const float* __restrict__ v6,
                       const float* __restrict__ Wl2, const float* __restrict__ bl2,
                       const float* __restrict__ g7, const float* __restrict__ b7,
                       const float* __restrict__ m7, const float* __restrict__ v7,
                       const float* __restrict__ Wl3, const float* __restrict__ bl3,
                       float* __restrict__ out) {
  const int b = blockIdx.x, tid = threadIdx.x;
  __shared__ float gb[1024], a1[512], a2[256];
  for (int i = tid; i < 1024; i += 256) gb[i] = g[(size_t)b * 1024 + i];
  __syncthreads();
  for (int o = tid; o < 512; o += 256) {
    float acc = 0.f;
    for (int c = 0; c < 1024; ++c) acc += Wl1[(size_t)o * 1024 + c] * gb[c];
    float sc = g6[o] * rsqrtf(v6[o] + EPSBN);
    float val = (acc - m6[o]) * sc + b6[o];
    a1[o] = (val > 0.f) ? val : 0.2f * val;
  }
  __syncthreads();
  for (int o = tid; o < 256; o += 256) {
    float acc = bl2[o];
    for (int c = 0; c < 512; ++c) acc += Wl2[(size_t)o * 512 + c] * a1[c];
    float sc = g7[o] * rsqrtf(v7[o] + EPSBN);
    float val = (acc - m7[o]) * sc + b7[o];
    a2[o] = (val > 0.f) ? val : 0.2f * val;
  }
  __syncthreads();
  if (tid < 40) {
    float acc = bl3[tid];
    for (int c = 0; c < 256; ++c) acc += Wl3[tid * 256 + c] * a2[c];
    out[b * 40 + tid] = acc;
  }
}

// ===========================================================================
extern "C" void kernel_launch(void* const* d_in, const int* in_sizes, int n_in,
                              void* d_out, int out_size, void* d_ws, size_t ws_size,
                              hipStream_t stream) {
  (void)in_sizes; (void)n_in; (void)out_size; (void)ws_size;
  auto F = [&](int i) { return (const float*)d_in[i]; };
  const float* x = F(0);

  char* w = (char*)d_ws;
  size_t off = 0;
  auto alloc = [&](size_t bytes) { void* p = w + off; off = (off + bytes + 255) & ~(size_t)255; return p; };
  float* pos  = (float*)alloc((size_t)16 * 3 * 1024 * 4);
  float* hres = (float*)alloc((size_t)16 * 1024 * 256 * 4);
  h16*   h0   = (h16*)  alloc((size_t)16 * 1024 * 256 * 2);
  h16*   cat  = (h16*)  alloc((size_t)16 * 1024 * 1024 * 2);
  h16*   xk   = (h16*)  alloc((size_t)16 * 1024 * 64 * 2);
  h16*   xv   = (h16*)  alloc((size_t)16 * 256 * 1024 * 2);
  h16*   xr   = (h16*)  alloc((size_t)16 * 1024 * 256 * 2);
  float* E    = (float*)alloc((size_t)1024 * 1024 * 4);
  h16*   PT   = (h16*)  alloc((size_t)1024 * 1024 * 2);
  float* rowM = (float*)alloc(1024 * 4);
  float* rowZi= (float*)alloc(1024 * 4);
  float* gbuf = (float*)alloc((size_t)16 * 1024 * 4);
  h16*   wkh  = (h16*)  alloc((size_t)4 * 16384 * 2);
  h16*   wvh  = (h16*)  alloc((size_t)4 * 65536 * 2);
  h16*   wth  = (h16*)  alloc((size_t)4 * 65536 * 2);
  h16*   wfh  = (h16*)  alloc((size_t)1048576 * 2);

  // f16 weight mirrors
  for (int L = 0; L < 4; ++L) {
    int sa = 16 + 9 * L;
    k_cvt<<<(16384 + 255) / 256, 256, 0, stream>>>(F(sa + 0), wkh + (size_t)L * 16384, 16384);
    k_cvt<<<(65536 + 255) / 256, 256, 0, stream>>>(F(sa + 1), wvh + (size_t)L * 65536, 65536);
    k_cvt<<<(65536 + 255) / 256, 256, 0, stream>>>(F(sa + 3), wth + (size_t)L * 65536, 65536);
  }
  k_cvt<<<(1048576 + 255) / 256, 256, 0, stream>>>(F(52), wfh, 1048576);

  // stem
  k_pos<<<48, 256, 0, stream>>>(x, pos);
  k_mlp0<<<dim3(16, 16), 256, 0, stream>>>(x, pos,
      F(1), F(2), F(3), F(4), F(5),
      F(6), F(7), F(8), F(9), F(10),
      F(11), F(12), F(13), F(14), F(15), hres, h0);

  // 4 self-attention layers
  for (int L = 0; L < 4; ++L) {
    int sa = 16 + 9 * L;
    const h16* inT = (L == 0) ? h0 : (cat + (size_t)(L - 1) * 256);
    int ldin = (L == 0) ? 256 : 1024;
    k_gemm_xk<<<dim3(16, 16), 256, 0, stream>>>(inT, ldin, wkh + (size_t)L * 16384, xk);
    k_gemm_xv<<<dim3(16, 4, 16), 256, 0, stream>>>(inT, ldin, wvh + (size_t)L * 65536,
                                                   F(sa + 2), xv);
    for (int b = 0; b < 16; ++b) {
      k_energy<<<dim3(16, 16), 256, 0, stream>>>(xk + (size_t)b * 1024 * 64, E);
      k_rowstat<<<128, 256, 0, stream>>>(E, rowM, rowZi);
      k_ptgen<<<4096, 256, 0, stream>>>(E, rowM, rowZi, PT);
      k_apply<<<dim3(16, 4), 256, 0, stream>>>(PT, xv + (size_t)b * 256 * 1024,
                                               xr + (size_t)b * 1024 * 256);
    }
    k_wt_res<<<dim3(16, 16), 256, 0, stream>>>(xr, wth + (size_t)L * 65536,
        F(sa + 4), F(sa + 5), F(sa + 6), F(sa + 7), F(sa + 8),
        hres, cat + (size_t)L * 256);
  }

  // fused feature conv + global max pool
  k_wf<<<dim3(16, 16), 256, 0, stream>>>(cat, wfh, F(53), F(54), F(55), F(56), gbuf);

  // head
  k_head<<<16, 256, 0, stream>>>(gbuf,
      F(57), F(58), F(59), F(60), F(61),
      F(62), F(63), F(64), F(65), F(66), F(67),
      F(68), F(69), (float*)d_out);
}
